// MPNetForPretraining_59657095741799
// MI455X (gfx1250) — matrix operand order repaired and checked by reading, not compiled, not run on hardware
//
#include <hip/hip_runtime.h>
#include <math.h>

// ---------------- problem constants ----------------
#define Bsz   4
#define Tlen  512
#define Pq    77
#define Vv    30522
#define Dd    768
#define Hh    12
#define FFd   3072
#define Ll    6
#define HDd   64
#define LcN   (Tlen - Pq)      // 435
#define SZn   (LcN - Pq)       // 358
#define LcPad 448              // padded key length (multiple of 32)
#define NEGF  (-1.0e9f)
#define SCALEF 0.125f          // 64^-0.5

// ---------------- small helpers ----------------
typedef __attribute__((ext_vector_type(16))) __bf16         bf16x16;
typedef __attribute__((ext_vector_type(8)))  float          f32x8;
typedef __attribute__((ext_vector_type(8)))  unsigned short u16x8;
typedef __attribute__((ext_vector_type(16))) unsigned short u16x16;

__device__ __forceinline__ unsigned short f2bf(float f) {
  unsigned int u = __builtin_bit_cast(unsigned int, f);
  u += 0x7FFFu + ((u >> 16) & 1u);          // round-to-nearest-even
  return (unsigned short)(u >> 16);
}

__device__ __forceinline__ bf16x16 load_frag(const unsigned short* p0,
                                             const unsigned short* p1) {
  u16x8 lo = *(const u16x8*)p0;
  u16x8 hi = *(const u16x8*)p1;
  u16x16 c = __builtin_shufflevector(lo, hi, 0,1,2,3,4,5,6,7,8,9,10,11,12,13,14,15);
  return __builtin_bit_cast(bf16x16, c);
}

// CDNA5 async global->LDS copy of one 16-byte chunk per active lane (ASYNCcnt).
__device__ __forceinline__ void async_b128(unsigned dsaddr, unsigned long long gaddr) {
  asm volatile("global_load_async_to_lds_b128 %0, %1, off"
               :: "v"(dsaddr), "v"(gaddr) : "memory");
}
__device__ __forceinline__ void wait_async0() {
  asm volatile("s_wait_asynccnt 0x0" ::: "memory");
}

// ---------------- generic bf16 WMMA GEMM ----------------
// C(M,N) = A(M,K)bf16 @ op(B)bf16 [+bias] [gelu] [+res], batched over grid.z.
// REQUIRES K % 32 == 0 (call sites pad K). OOB M/N rows are *clamped*, not
// zero-filled: their outputs are never stored, so no zero-fill or per-element
// predication is needed -> branch-free main loop, every chunk is one
// GLOBAL_LOAD_ASYNC_TO_LDS_B128. Double-buffered LDS, one barrier per K-step.
// Block tile 128x64, 8 waves, 32x32 per wave -> 4 WMMAs per wave per K-step.
#define LDS_S 40
template<bool TRANSB>
__global__ void __launch_bounds__(256)
gemm_bf16(const unsigned short* __restrict__ A, int lda, long sAb, long sAh,
          const unsigned short* __restrict__ Bm, int ldb, long sBb, long sBh,
          const float* __restrict__ bias,
          const float* __restrict__ res, int ldr,
          float* __restrict__ Cf, unsigned short* __restrict__ Cb,
          int ldc, long sCb, long sCh,
          int M, int N, int K, int Hdim, int gelu_flag)
{
  __shared__ __align__(16) unsigned short lA[2][128 * LDS_S];
  __shared__ __align__(16) unsigned short lB[2][64 * LDS_S];   // [n][k]

  const int z  = blockIdx.z;
  const int bb = z / Hdim, hh = z - bb * Hdim;
  A  += bb * sAb + hh * sAh;
  Bm += bb * sBb + hh * sBh;
  const long coff = bb * sCb + hh * sCh;
  if (Cf) Cf += coff;
  if (Cb) Cb += coff;

  const int bm = blockIdx.y * 128, bn = blockIdx.x * 64;
  const int t = threadIdx.x;
  const int w = t >> 5, lane = t & 31;
  const int wmRow = (w >> 1) * 32, wnCol = (w & 1) * 32;
  const int hl = lane & 15, sel = lane >> 4;

  // per-thread chunk coordinates (row-clamped pointers, unconditional loads)
  const int rA = t >> 2, cA = (t & 3) * 8;           // A: 128 rows x 32 k (2 chunks)
  int a0r = bm + rA;        if (a0r > M - 1) a0r = M - 1;
  int a1r = bm + rA + 64;   if (a1r > M - 1) a1r = M - 1;
  const unsigned short* aBase0 = A + (long)a0r * lda + cA;
  const unsigned short* aBase1 = A + (long)a1r * lda + cA;
  // TRANSB B: 64 n-rows x 32 k (1 chunk); else B: 32 k x 64 n (vector+scatter)
  const int rB = t >> 2, cB = (t & 3) * 8;
  int bnr = bn + rB;        if (bnr > N - 1) bnr = N - 1;
  const unsigned short* bBase = Bm + (long)bnr * ldb + cB;
  const int kB = t >> 3, nB = (t & 7) * 8;

  auto stage = [&](int k0, int buf) {
    async_b128((unsigned)(size_t)&lA[buf][rA * LDS_S + cA],
               (unsigned long long)(size_t)(aBase0 + k0));
    async_b128((unsigned)(size_t)&lA[buf][(rA + 64) * LDS_S + cA],
               (unsigned long long)(size_t)(aBase1 + k0));
    if (TRANSB) {
      async_b128((unsigned)(size_t)&lB[buf][rB * LDS_S + cB],
                 (unsigned long long)(size_t)(bBase + k0));
    } else {
      const unsigned short* src = Bm + (long)(k0 + kB) * ldb + bn + nB;
      u16x8 v = *(const u16x8*)src;
      #pragma unroll
      for (int j = 0; j < 8; ++j) lB[buf][(nB + j) * LDS_S + kB] = v[j];
    }
  };

  f32x8 acc00 = {}, acc01 = {}, acc10 = {}, acc11 = {};
  stage(0, 0);
  int cur = 0;
  for (int k0 = 0; k0 < K; k0 += 32) {
    wait_async0();
    __syncthreads();                    // buf[cur] ready; prior reads retired
    if (k0 + 32 < K) stage(k0 + 32, cur ^ 1);

    // B frags: lane = col, K span sel*16..+16 (2 contiguous b128 each)
    const unsigned short* bp0 = &lB[cur][(wnCol + hl) * LDS_S + sel * 16];
    const unsigned short* bp1 = &lB[cur][(wnCol + 16 + hl) * LDS_S + sel * 16];
    bf16x16 b0 = load_frag(bp0, bp0 + 8);
    bf16x16 b1 = load_frag(bp1, bp1 + 8);
    // A frags: lanes 0-15 K {0..15}, lanes 16-31 K {16..31}
    const unsigned short* ap0 = &lA[cur][(wmRow + hl) * LDS_S + sel * 16];
    const unsigned short* ap1 = &lA[cur][(wmRow + 16 + hl) * LDS_S + sel * 16];
    bf16x16 a0 = load_frag(ap0, ap0 + 8);
    bf16x16 a1 = load_frag(ap1, ap1 + 8);
    acc00 = __builtin_amdgcn_wmma_f32_16x16x32_bf16(false, a0, false, b0, (short)0, acc00, false, false);
    acc01 = __builtin_amdgcn_wmma_f32_16x16x32_bf16(false, a0, false, b1, (short)0, acc01, false, false);
    acc10 = __builtin_amdgcn_wmma_f32_16x16x32_bf16(false, a1, false, b0, (short)0, acc10, false, false);
    acc11 = __builtin_amdgcn_wmma_f32_16x16x32_bf16(false, a1, false, b1, (short)0, acc11, false, false);
    cur ^= 1;
  }

  // epilogue: C layout — VGPR r: lanes 0-15 row r, lanes 16-31 row r+8
  #pragma unroll
  for (int ct = 0; ct < 2; ++ct) {
    const int col = bn + wnCol + ct * 16 + hl;
    const float bcol = (bias && col < N) ? bias[col] : 0.0f;
    #pragma unroll
    for (int rt = 0; rt < 2; ++rt) {
      f32x8 acc = ct ? (rt ? acc11 : acc01) : (rt ? acc10 : acc00);
      const int rbase = bm + wmRow + rt * 16 + sel * 8;
      #pragma unroll
      for (int r = 0; r < 8; ++r) {
        const int row = rbase + r;
        if (row < M && col < N) {
          float v = acc[r] + bcol;
          if (gelu_flag) v = 0.5f * v * (1.0f + erff(v * 0.70710678118654752f));
          if (res) v += res[(long)row * ldr + col];
          if (Cf) Cf[(long)row * ldc + col] = v;
          if (Cb) Cb[(long)row * ldc + col] = f2bf(v);
        }
      }
    }
  }
}

// ---------------- elementwise / reduction kernels ----------------
__global__ void __launch_bounds__(256)
cvt_kernel(const float* __restrict__ src, unsigned short* __restrict__ dst, long n) {
  long i = (long)blockIdx.x * 256 + threadIdx.x;
  if (i < n) dst[i] = f2bf(src[i]);
}

__global__ void __launch_bounds__(256)
zero_kernel(unsigned short* __restrict__ p, long n) {
  long i = (long)blockIdx.x * 256 + threadIdx.x;
  if (i < n) p[i] = 0;
}

// fp32 (R x C) -> bf16 transposed (C x R), LDS-tiled 32x32
__global__ void __launch_bounds__(256)
cvtT_kernel(const float* __restrict__ src, unsigned short* __restrict__ dst,
            int R, int C) {
  __shared__ float tile[32][33];
  const int bc = blockIdx.x * 32, br = blockIdx.y * 32;
  const int tx = threadIdx.x & 31, ty = threadIdx.x >> 5;
  #pragma unroll
  for (int i = 0; i < 32; i += 8) {
    int r = br + ty + i, c = bc + tx;
    tile[ty + i][tx] = (r < R && c < C) ? src[(long)r * C + c] : 0.f;
  }
  __syncthreads();
  #pragma unroll
  for (int i = 0; i < 32; i += 8) {
    int c = bc + ty + i, r = br + tx;
    if (c < C && r < R) dst[(long)c * R + r] = f2bf(tile[tx][ty + i]);
  }
}

__global__ void __launch_bounds__(256)
embed_kernel(const float* __restrict__ tok, const float* __restrict__ pos,
             const int* __restrict__ ids, const int* __restrict__ poss,
             float* __restrict__ cbuf, float* __restrict__ qbuf) {
  long idx = (long)blockIdx.x * 256 + threadIdx.x;
  if (idx >= (long)Bsz * Tlen * Dd) return;
  int d = idx % Dd;
  long rt = idx / Dd;
  int b = rt / Tlen, tt = rt % Tlen;
  float v = tok[(long)ids[b * Tlen + tt] * Dd + d] +
            pos[(long)poss[b * Tlen + tt] * Dd + d];
  if (tt < LcN) cbuf[((long)b * LcN + tt) * Dd + d] = v;
  else          qbuf[((long)b * Pq + (tt - LcN)) * Dd + d] = v;
}

__global__ void __launch_bounds__(256)
bucket_kernel(const int* __restrict__ poss, int* __restrict__ bucket) {
  long idx = (long)blockIdx.x * 256 + threadIdx.x;
  if (idx >= (long)Bsz * LcN * LcN) return;
  int b = idx / (LcN * LcN);
  int rem = idx % (LcN * LcN);
  int i = rem / LcN, j = rem % LcN;
  int rp = poss[b * Tlen + j] - poss[b * Tlen + i];
  int n = -rp;
  int ret = (n < 0) ? 16 : 0;
  n = (n < 0) ? -n : n;
  int out;
  if (n < 8) out = n;
  else {
    float nl = (float)(n < 1 ? 1 : n);
    int large = 8 + (int)(logf(nl * 0.125f) / logf(16.0f) * 8.0f);
    out = large > 15 ? 15 : large;
  }
  bucket[idx] = ret + out;
}

__global__ void __launch_bounds__(256)
ln_kernel(const float* __restrict__ x, const float* __restrict__ g,
          const float* __restrict__ be, unsigned short* __restrict__ out) {
  __shared__ float red[2][8];
  long row = blockIdx.x;
  const float* xr = x + row * Dd;
  float s = 0.f, sq = 0.f;
  for (int d = threadIdx.x; d < Dd; d += 256) { float v = xr[d]; s += v; sq += v * v; }
  for (int o = 16; o > 0; o >>= 1) { s += __shfl_down(s, o, 32); sq += __shfl_down(sq, o, 32); }
  int w = threadIdx.x >> 5, lane = threadIdx.x & 31;
  if (lane == 0) { red[0][w] = s; red[1][w] = sq; }
  __syncthreads();
  if (threadIdx.x == 0) {
    float ts = 0.f, tq = 0.f;
    for (int i = 0; i < 8; ++i) { ts += red[0][i]; tq += red[1][i]; }
    red[0][0] = ts; red[1][0] = tq;
  }
  __syncthreads();
  float mean = red[0][0] * (1.0f / Dd);
  float var  = red[1][0] * (1.0f / Dd) - mean * mean;
  float inv  = rsqrtf(var + 1e-5f);
  unsigned short* orow = out + row * Dd;
  for (int d = threadIdx.x; d < Dd; d += 256)
    orow[d] = f2bf((xr[d] - mean) * inv * g[d] + be[d]);
}

// S row: scale, add rel-bias via bucket lookup, apply analytic mask, pad with NEG
__global__ void __launch_bounds__(256)
biasmask_kernel(float* __restrict__ S, const int* __restrict__ bucket,
                const float* __restrict__ relb, int Mq, int qmode) {
  long z = blockIdx.x;
  int b = z / (Hh * Mq);
  int rem = z % (Hh * Mq);
  int h = rem / Mq, r = rem % Mq;
  int crow = qmode ? (SZn + r) : r;
  float* srow = S + z * (long)LcPad;
  const int* brow = bucket + ((long)b * LcN + crow) * LcN;
  for (int c = threadIdx.x; c < LcPad; c += 256) {
    float v;
    if (c >= LcN) v = NEGF;
    else {
      bool allowed;
      if (qmode) {
        allowed = (c < SZn) || ((c - SZn) <= r);
      } else {
        if (c < SZn - Pq) allowed = true;
        else if (c < SZn) { int j = c - (SZn - Pq); allowed = (r >= SZn) && (j <= r - SZn); }
        else              { int j = c - SZn;        allowed = !((r >= SZn) && (j <= r - SZn)); }
      }
      v = allowed ? (SCALEF * srow[c] + relb[brow[c] * Hh + h]) : NEGF;
    }
    srow[c] = v;
  }
}

__global__ void __launch_bounds__(128)
softmax_kernel(const float* __restrict__ S, unsigned short* __restrict__ Pr) {
  __shared__ float red[4];
  long row = blockIdx.x;
  const float* sr = S + row * (long)LcPad;
  int w = threadIdx.x >> 5, lane = threadIdx.x & 31;
  float mx = -3.4e38f;
  for (int c = threadIdx.x; c < LcPad; c += 128) mx = fmaxf(mx, sr[c]);
  for (int o = 16; o > 0; o >>= 1) mx = fmaxf(mx, __shfl_down(mx, o, 32));
  if (lane == 0) red[w] = mx;
  __syncthreads();
  if (threadIdx.x == 0) {
    float m = red[0];
    for (int i = 1; i < 4; ++i) m = fmaxf(m, red[i]);
    red[0] = m;
  }
  __syncthreads();
  mx = red[0];
  __syncthreads();
  float s = 0.f;
  for (int c = threadIdx.x; c < LcPad; c += 128) s += expf(sr[c] - mx);
  for (int o = 16; o > 0; o >>= 1) s += __shfl_down(s, o, 32);
  if (lane == 0) red[w] = s;
  __syncthreads();
  if (threadIdx.x == 0) { float t = 0.f; for (int i = 0; i < 4; ++i) t += red[i]; red[0] = t; }
  __syncthreads();
  float inv = 1.0f / red[0];
  unsigned short* pr = Pr + row * (long)LcPad;
  for (int c = threadIdx.x; c < LcPad; c += 128) pr[c] = f2bf(expf(sr[c] - mx) * inv);
}

// ---------------- host orchestration ----------------
extern "C" void kernel_launch(void* const* d_in, const int* in_sizes, int n_in,
                              void* d_out, int out_size, void* d_ws, size_t ws_size,
                              hipStream_t stream) {
  (void)in_sizes; (void)n_in; (void)out_size; (void)ws_size;
  const float* tok_emb  = (const float*)d_in[0];
  const float* pos_emb  = (const float*)d_in[1];
  const float* rel_bias = (const float*)d_in[2];
  const float* ln1_s = (const float*)d_in[3];
  const float* ln1_b = (const float*)d_in[4];
  const float* wq = (const float*)d_in[5];
  const float* bq = (const float*)d_in[6];
  const float* wk = (const float*)d_in[7];
  const float* bk = (const float*)d_in[8];
  const float* wv = (const float*)d_in[9];
  const float* bv = (const float*)d_in[10];
  const float* wo = (const float*)d_in[11];
  const float* bo = (const float*)d_in[12];
  const float* ln2_s = (const float*)d_in[13];
  const float* ln2_b = (const float*)d_in[14];
  const float* w1 = (const float*)d_in[15];
  const float* b1 = (const float*)d_in[16];
  const float* w2 = (const float*)d_in[17];
  const float* b2 = (const float*)d_in[18];
  const float* fin_s = (const float*)d_in[19];
  const float* fin_b = (const float*)d_in[20];
  const float* head_w = (const float*)d_in[21];
  const float* head_b = (const float*)d_in[22];
  const float* hln_s = (const float*)d_in[23];
  const float* hln_b = (const float*)d_in[24];
  const float* head_bias = (const float*)d_in[25];
  const int* input_ids = (const int*)d_in[26];
  const int* positions = (const int*)d_in[27];
  float* out = (float*)d_out;

  // ---- workspace bump allocator ----
  char* wsB = (char*)d_ws;
  size_t off = 0;
  auto alloc = [&](size_t bytes) -> void* {
    void* p = wsB + off;
    off = (off + bytes + 255) & ~(size_t)255;
    return p;
  };
  const long MC = (long)Bsz * LcN;   // 1740 context rows
  const long MQ = (long)Bsz * Pq;    // 308 query rows
  const int  VPAD = 16;              // zeroed pad rows after V (K padded to 448)

  // all weights stored bf16 TRANSPOSED (out x in) so GEMMs take the async path
  unsigned short* wq_b = (unsigned short*)alloc((size_t)Ll * Dd * Dd * 2);
  unsigned short* wk_b = (unsigned short*)alloc((size_t)Ll * Dd * Dd * 2);
  unsigned short* wv_b = (unsigned short*)alloc((size_t)Ll * Dd * Dd * 2);
  unsigned short* wo_b = (unsigned short*)alloc((size_t)Ll * Dd * Dd * 2);
  unsigned short* w1_b = (unsigned short*)alloc((size_t)Ll * Dd * FFd * 2);
  unsigned short* w2_b = (unsigned short*)alloc((size_t)Ll * FFd * Dd * 2);
  unsigned short* hw_b = (unsigned short*)alloc((size_t)Dd * Dd * 2);
  unsigned short* tok_b = (unsigned short*)alloc((size_t)Vv * Dd * 2); // natural (V x D)
  float* c_f32 = (float*)alloc((size_t)MC * Dd * 4);
  float* q_f32 = (float*)alloc((size_t)MQ * Dd * 4);
  unsigned short* c_ln = (unsigned short*)alloc((size_t)MC * Dd * 2);
  unsigned short* q_ln = (unsigned short*)alloc((size_t)MQ * Dd * 2);
  unsigned short* k_b  = (unsigned short*)alloc((size_t)MC * Dd * 2);
  unsigned short* v_b  = (unsigned short*)alloc((size_t)(MC + VPAD) * Dd * 2);
  unsigned short* qc_b = (unsigned short*)alloc((size_t)MC * Dd * 2);
  unsigned short* qq_b = (unsigned short*)alloc((size_t)MQ * Dd * 2);
  unsigned short* oc_b = (unsigned short*)alloc((size_t)MC * Dd * 2);
  unsigned short* oq_b = (unsigned short*)alloc((size_t)MQ * Dd * 2);
  float* sc_c = (float*)alloc((size_t)Bsz * Hh * LcN * LcPad * 4);
  float* sc_q = (float*)alloc((size_t)Bsz * Hh * Pq * LcPad * 4);
  unsigned short* pr_c = (unsigned short*)alloc((size_t)Bsz * Hh * LcN * LcPad * 2);
  unsigned short* pr_q = (unsigned short*)alloc((size_t)Bsz * Hh * Pq * LcPad * 2);
  unsigned short* ff_c = (unsigned short*)alloc((size_t)MC * FFd * 2);
  unsigned short* ff_q = (unsigned short*)alloc((size_t)MQ * FFd * 2);
  int* bucket = (int*)alloc((size_t)Bsz * LcN * LcN * 4);
  float* h_f32 = (float*)alloc((size_t)MQ * Dd * 4);
  unsigned short* hn_b = (unsigned short*)alloc((size_t)MQ * Dd * 2);

  auto cvt = [&](const float* s, unsigned short* d, long n) {
    cvt_kernel<<<dim3((unsigned)((n + 255) / 256)), dim3(256), 0, stream>>>(s, d, n);
  };
  auto cvtT = [&](const float* s, unsigned short* d, int R, int C) {
    cvtT_kernel<<<dim3((C + 31) / 32, (R + 31) / 32), dim3(256), 0, stream>>>(s, d, R, C);
  };
  auto gemm = [&](bool transB,
                  const unsigned short* A, int lda, long sAb, long sAh,
                  const unsigned short* Bm, int ldb, long sBb, long sBh,
                  const float* bias, const float* res, int ldr,
                  float* Cf, unsigned short* Cb, int ldc, long sCb, long sCh,
                  int M, int N, int K, int Hdim, int batches, int gelu) {
    dim3 g((N + 63) / 64, (M + 127) / 128, batches), blk(256);
    if (transB)
      gemm_bf16<true><<<g, blk, 0, stream>>>(A, lda, sAb, sAh, Bm, ldb, sBb, sBh,
          bias, res, ldr, Cf, Cb, ldc, sCb, sCh, M, N, K, Hdim, gelu);
    else
      gemm_bf16<false><<<g, blk, 0, stream>>>(A, lda, sAb, sAh, Bm, ldb, sBb, sBh,
          bias, res, ldr, Cf, Cb, ldc, sCb, sCh, M, N, K, Hdim, gelu);
  };

  // ---- weight conversion: fp32 -> bf16, transposed to (out x in) ----
  for (int l = 0; l < Ll; ++l) {
    cvtT(wq + (long)l * Dd * Dd,  wq_b + (long)l * Dd * Dd,  Dd, Dd);
    cvtT(wk + (long)l * Dd * Dd,  wk_b + (long)l * Dd * Dd,  Dd, Dd);
    cvtT(wv + (long)l * Dd * Dd,  wv_b + (long)l * Dd * Dd,  Dd, Dd);
    cvtT(wo + (long)l * Dd * Dd,  wo_b + (long)l * Dd * Dd,  Dd, Dd);
    cvtT(w1 + (long)l * Dd * FFd, w1_b + (long)l * Dd * FFd, Dd, FFd);   // -> (FF x D)
    cvtT(w2 + (long)l * FFd * Dd, w2_b + (long)l * FFd * Dd, FFd, Dd);   // -> (D x FF)
  }
  cvtT(head_w, hw_b, Dd, Dd);
  cvt(tok_emb, tok_b, (long)Vv * Dd);       // logits use tok_emb^T == natural (V x D)
  // zero the V pad rows once per call (K-overrun in PV reads finite zeros)
  zero_kernel<<<dim3((unsigned)((VPAD * Dd + 255) / 256)), dim3(256), 0, stream>>>(
      v_b + MC * Dd, (long)VPAD * Dd);

  // ---- embedding + rel-position buckets ----
  {
    long n = (long)Bsz * Tlen * Dd;
    embed_kernel<<<dim3((unsigned)((n + 255) / 256)), dim3(256), 0, stream>>>(
        tok_emb, pos_emb, input_ids, positions, c_f32, q_f32);
    long nb = (long)Bsz * LcN * LcN;
    bucket_kernel<<<dim3((unsigned)((nb + 255) / 256)), dim3(256), 0, stream>>>(
        positions, bucket);
  }

  const long sHB = (long)Hh * LcN * LcPad, sHH = (long)LcN * LcPad;
  const long sQB = (long)Hh * Pq * LcPad,  sQH = (long)Pq * LcPad;

  for (int l = 0; l < Ll; ++l) {
    const unsigned short* Wq = wq_b + (long)l * Dd * Dd;
    const unsigned short* Wk = wk_b + (long)l * Dd * Dd;
    const unsigned short* Wv = wv_b + (long)l * Dd * Dd;
    const unsigned short* Wo = wo_b + (long)l * Dd * Dd;
    const unsigned short* W1 = w1_b + (long)l * Dd * FFd;   // (FF x D)
    const unsigned short* W2 = w2_b + (long)l * FFd * Dd;   // (D x FF)
    const float* Bq = bq + (long)l * Dd;  const float* Bk = bk + (long)l * Dd;
    const float* Bv = bv + (long)l * Dd;  const float* Bo = bo + (long)l * Dd;
    const float* B1 = b1 + (long)l * FFd; const float* B2 = b2 + (long)l * Dd;
    const float* G1 = ln1_s + (long)l * Dd; const float* E1 = ln1_b + (long)l * Dd;
    const float* G2 = ln2_s + (long)l * Dd; const float* E2 = ln2_b + (long)l * Dd;

    // LN1
    ln_kernel<<<dim3((unsigned)MC), dim3(256), 0, stream>>>(c_f32, G1, E1, c_ln);
    ln_kernel<<<dim3((unsigned)MQ), dim3(256), 0, stream>>>(q_f32, G1, E1, q_ln);

    // projections (transposed weights -> TRANSB/async path)
    gemm(true, c_ln, Dd,0,0, Wk, Dd,0,0, Bk, nullptr,0, nullptr, k_b,  Dd,0,0, MC, Dd, Dd, 1, 1, 0);
    gemm(true, c_ln, Dd,0,0, Wv, Dd,0,0, Bv, nullptr,0, nullptr, v_b,  Dd,0,0, MC, Dd, Dd, 1, 1, 0);
    gemm(true, c_ln, Dd,0,0, Wq, Dd,0,0, Bq, nullptr,0, nullptr, qc_b, Dd,0,0, MC, Dd, Dd, 1, 1, 0);
    gemm(true, q_ln, Dd,0,0, Wq, Dd,0,0, Bq, nullptr,0, nullptr, qq_b, Dd,0,0, MQ, Dd, Dd, 1, 1, 0);

    // S = Q K^T, batched over (b,h); scale+bias+mask fused afterwards
    gemm(true, qc_b, Dd, (long)LcN*Dd, HDd, k_b, Dd, (long)LcN*Dd, HDd,
         nullptr, nullptr,0, sc_c, nullptr, LcPad, sHB, sHH,
         LcN, LcN, HDd, Hh, Bsz*Hh, 0);
    gemm(true, qq_b, Dd, (long)Pq*Dd, HDd, k_b, Dd, (long)LcN*Dd, HDd,
         nullptr, nullptr,0, sc_q, nullptr, LcPad, sQB, sQH,
         Pq, LcN, HDd, Hh, Bsz*Hh, 0);
    biasmask_kernel<<<dim3((unsigned)(Bsz*Hh*LcN)), dim3(256), 0, stream>>>(sc_c, bucket, rel_bias, LcN, 0);
    biasmask_kernel<<<dim3((unsigned)(Bsz*Hh*Pq)),  dim3(256), 0, stream>>>(sc_q, bucket, rel_bias, Pq, 1);
    softmax_kernel<<<dim3((unsigned)(Bsz*Hh*LcN)), dim3(128), 0, stream>>>(sc_c, pr_c);
    softmax_kernel<<<dim3((unsigned)(Bsz*Hh*Pq)),  dim3(128), 0, stream>>>(sc_q, pr_q);

    // O = P V, batched, K padded to 448 (probs cols 435..447 are exact zeros)
    gemm(false, pr_c, LcPad, sHB, sHH, v_b, Dd, (long)LcN*Dd, HDd,
         nullptr, nullptr,0, nullptr, oc_b, Dd, (long)LcN*Dd, HDd,
         LcN, HDd, LcPad, Hh, Bsz*Hh, 0);
    gemm(false, pr_q, LcPad, sQB, sQH, v_b, Dd, (long)LcN*Dd, HDd,
         nullptr, nullptr,0, nullptr, oq_b, Dd, (long)Pq*Dd, HDd,
         Pq, HDd, LcPad, Hh, Bsz*Hh, 0);

    // output projection + residual (in place on the fp32 residual stream)
    gemm(true, oc_b, Dd,0,0, Wo, Dd,0,0, Bo, c_f32, Dd, c_f32, nullptr, Dd,0,0, MC, Dd, Dd, 1, 1, 0);
    gemm(true, oq_b, Dd,0,0, Wo, Dd,0,0, Bo, q_f32, Dd, q_f32, nullptr, Dd,0,0, MQ, Dd, Dd, 1, 1, 0);

    // FFN: LN2 -> GELU(W1) -> W2 + residual
    ln_kernel<<<dim3((unsigned)MC), dim3(256), 0, stream>>>(c_f32, G2, E2, c_ln);
    ln_kernel<<<dim3((unsigned)MQ), dim3(256), 0, stream>>>(q_f32, G2, E2, q_ln);
    gemm(true, c_ln, Dd,0,0, W1, Dd,0,0, B1, nullptr,0, nullptr, ff_c, FFd,0,0, MC, FFd, Dd, 1, 1, 1);
    gemm(true, q_ln, Dd,0,0, W1, Dd,0,0, B1, nullptr,0, nullptr, ff_q, FFd,0,0, MQ, FFd, Dd, 1, 1, 1);
    gemm(true, ff_c, FFd,0,0, W2, FFd,0,0, B2, c_f32, Dd, c_f32, nullptr, Dd,0,0, MC, Dd, FFd, 1, 1, 0);
    gemm(true, ff_q, FFd,0,0, W2, FFd,0,0, B2, q_f32, Dd, q_f32, nullptr, Dd,0,0, MQ, Dd, FFd, 1, 1, 0);
  }

  // ---- head: final LN -> GELU dense -> LN -> vocab GEMM (tok_emb natural N x K) ----
  ln_kernel<<<dim3((unsigned)MQ), dim3(256), 0, stream>>>(q_f32, fin_s, fin_b, q_ln);
  gemm(true, q_ln, Dd,0,0, hw_b, Dd,0,0, head_b, nullptr,0, h_f32, nullptr, Dd,0,0, MQ, Dd, Dd, 1, 1, 1);
  ln_kernel<<<dim3((unsigned)MQ), dim3(256), 0, stream>>>(h_f32, hln_s, hln_b, hn_b);
  gemm(true, hn_b, Dd,0,0, tok_b, Dd,0,0, head_bias, nullptr,0,
       out, nullptr, Vv,0,0, (int)MQ, Vv, Dd, 1, 1, 0);
}
